// Trasnsformer_model_56126632624349
// MI455X (gfx1250) — matrix operand order repaired and checked by reading, not compiled, not run on hardware
//
#include <hip/hip_runtime.h>
#include <hip/hip_bf16.h>

// ---------------------------------------------------------------------------
// CDNA5 (gfx1250, wave32) WMMA transformer block.  E=128, H=8, L=2048.
// All GEMMs via v_wmma_f32_16x16x32_bf16; all fragment loads/stores are
// 128-bit vector ops thanks to transposed (N-major, K-contiguous) layouts
// for every B-side operand.
// ---------------------------------------------------------------------------

typedef __attribute__((ext_vector_type(16))) __bf16 v16bf;
typedef __attribute__((ext_vector_type(8)))  __bf16 v8bf;
typedef __attribute__((ext_vector_type(8)))  float  v8f;

#define E_DIM  128
#define H_NUM  8
#define L_SEQ  2048
#define NITEMS 50000
#define WAMT   5
#define EH     (E_DIM * H_NUM)   // 1024

__device__ __forceinline__ v8f wmma_bf16(v16bf a, v16bf b, v8f c) {
    return __builtin_amdgcn_wmma_f32_16x16x32_bf16(
        false, a, false, b, (short)0, c, false, false);
}

// A fragment, row-major source with K contiguous:
//   A[m,k] = src[(m0+m)*lda + k0+k].  Per lane: two contiguous 16B chunks
//   (K = half*8..+7 and 16+half*8..+7), matching the ISA 16-bit A layout.
__device__ __forceinline__ v16bf load_a_rm(const __bf16* src, int lda,
                                           int m0, int k0, int lane) {
    int half = lane >> 4, m = lane & 15;
    const __bf16* p = src + (size_t)(m0 + m) * lda + k0 + half * 8;
    v8bf lo = *(const v8bf*)p;          // K = half*8 .. half*8+7
    v8bf hi = *(const v8bf*)(p + 16);   // K = 16+half*8 ..
    return __builtin_shufflevector(lo, hi,
        0,1,2,3,4,5,6,7,8,9,10,11,12,13,14,15);
}

// B fragment from an N-major (transposed) source with K contiguous:
//   B[k,n] = src[(n0+n)*ldk + k0+k].  Per lane: two contiguous 16B chunks
//   (K = half*8..+7 and 16+half*8..+7), matching the A fragment layout.
__device__ __forceinline__ v16bf load_b_t(const __bf16* src, int ldk,
                                          int k0, int n0, int lane) {
    int half = lane >> 4, n = lane & 15;
    const __bf16* p = src + (size_t)(n0 + n) * ldk + k0 + half * 8;
    v8bf lo = *(const v8bf*)p;
    v8bf hi = *(const v8bf*)(p + 16);
    return __builtin_shufflevector(lo, hi,
        0,1,2,3,4,5,6,7,8,9,10,11,12,13,14,15);
}

// Convert a C/D fragment's 8 f32 to bf16 and store as one 16B vector
// (the fragment's 8 rows m = r+8*half are contiguous in our layouts).
__device__ __forceinline__ void store_bf8(__bf16* dst, v8f acc) {
    v8bf r;
#pragma unroll
    for (int i = 0; i < 8; ++i) r[i] = (__bf16)acc[i];
    *(v8bf*)dst = r;
}

// ---------------------------------------------------------------------------
// Kernel 1: f32 -> bf16 conversion
// ---------------------------------------------------------------------------
__global__ void cvt_f32_bf16_kernel(const float* __restrict__ s,
                                    __bf16* __restrict__ d, int n) {
    int i = blockIdx.x * blockDim.x + threadIdx.x;
    if (i < n) d[i] = (__bf16)s[i];
}

// ---------------------------------------------------------------------------
// Kernel 2: gather I = item[:,sq] + wrep[:,cq] into transposed layout
// I_t (L x 128, e contiguous); emit I_q.T (out2) and u_state row 0.
// grid = (L_SEQ), block = (E_DIM)
// ---------------------------------------------------------------------------
__global__ void gather_kernel(const int* __restrict__ sq,
                              const int* __restrict__ cq,
                              const float* __restrict__ item,
                              const float* __restrict__ wrep,
                              const float* __restrict__ uinit,
                              __bf16* __restrict__ I_t,
                              float* __restrict__ out1,
                              float* __restrict__ out2) {
    int l = blockIdx.x;
    int e = threadIdx.x;
    int s = sq[l];
    int c = cq[l];
    float iq = item[e * NITEMS + s];
    float iw = wrep[e * WAMT + c];
    I_t[l * E_DIM + e] = (__bf16)(iq + iw);
    out2[l * E_DIM + e] = iq;            // I_q.T
    if (l == 0) out1[e] = uinit[e];      // u_state row 0 = user_init
}

// ---------------------------------------------------------------------------
// Kernel 3: QKV projections.
//   z=0/1:  Q^T,K^T (L x 1024)  =  (W @ I)^T   -- A=W row-major, B=I_t
//   z=2  :  V (1024 x L row-major) via V^T = I^T @ W_v^T -- A=I_t, B=W_v
// block = 256 (8 waves, 32x32 tile each => 64x128 block tile)
// grid = (16, 16, 3)
// ---------------------------------------------------------------------------
__global__ void gemm_qkv_kernel(const __bf16* __restrict__ I_t,
                                const __bf16* __restrict__ Wq,
                                const __bf16* __restrict__ Wk,
                                const __bf16* __restrict__ Wv,
                                __bf16* __restrict__ Qt,
                                __bf16* __restrict__ Kt,
                                __bf16* __restrict__ Vr) {
    const int lane = threadIdx.x & 31, w = threadIdx.x >> 5;
    const int wm = w >> 2, wn = w & 3;
    const int half = lane >> 4, n = lane & 15;

    if (blockIdx.z < 2) {
        const __bf16* W = blockIdx.z ? Wk : Wq;
        __bf16* Ot = blockIdx.z ? Kt : Qt;
        const int m0 = blockIdx.x * 64 + wm * 32;    // weight row (0..1023)
        const int n0 = blockIdx.y * 128 + wn * 32;   // seq col (0..2047)
        v8f acc[2][2] = {};
#pragma unroll
        for (int k0 = 0; k0 < E_DIM; k0 += 32) {
            v16bf a0 = load_a_rm(W, E_DIM, m0,      k0, lane);
            v16bf a1 = load_a_rm(W, E_DIM, m0 + 16, k0, lane);
            v16bf b0 = load_b_t(I_t, E_DIM, k0, n0,      lane);
            v16bf b1 = load_b_t(I_t, E_DIM, k0, n0 + 16, lane);
            acc[0][0] = wmma_bf16(a0, b0, acc[0][0]);
            acc[0][1] = wmma_bf16(a0, b1, acc[0][1]);
            acc[1][0] = wmma_bf16(a1, b0, acc[1][0]);
            acc[1][1] = wmma_bf16(a1, b1, acc[1][1]);
        }
        // transposed store: Ot[(seq)*1024 + m], m contiguous per lane
#pragma unroll
        for (int mi = 0; mi < 2; ++mi)
#pragma unroll
            for (int ni = 0; ni < 2; ++ni)
                store_bf8(&Ot[(size_t)(n0 + ni * 16 + n) * EH +
                              m0 + mi * 16 + 8 * half], acc[mi][ni]);
    } else {
        // V^T = I^T @ W_v^T : M = L (seq), N = 1024 (v-row), K = 128
        const int flat = blockIdx.y * gridDim.x + blockIdx.x;  // 0..255
        const int m0 = (flat & 31) * 64 + wm * 32;    // seq (0..2047)
        const int n0 = (flat >> 5) * 128 + wn * 32;   // v-row (0..1023)
        v8f acc[2][2] = {};
#pragma unroll
        for (int k0 = 0; k0 < E_DIM; k0 += 32) {
            v16bf a0 = load_a_rm(I_t, E_DIM, m0,      k0, lane);
            v16bf a1 = load_a_rm(I_t, E_DIM, m0 + 16, k0, lane);
            v16bf b0 = load_b_t(Wv, E_DIM, k0, n0,      lane);   // W_v row-major == N-major/K-contig
            v16bf b1 = load_b_t(Wv, E_DIM, k0, n0 + 16, lane);
            acc[0][0] = wmma_bf16(a0, b0, acc[0][0]);
            acc[0][1] = wmma_bf16(a0, b1, acc[0][1]);
            acc[1][0] = wmma_bf16(a1, b0, acc[1][0]);
            acc[1][1] = wmma_bf16(a1, b1, acc[1][1]);
        }
        // store V row-major: Vr[(vrow)*L + seq], seq contiguous per lane
#pragma unroll
        for (int mi = 0; mi < 2; ++mi)
#pragma unroll
            for (int ni = 0; ni < 2; ++ni)
                store_bf8(&Vr[(size_t)(n0 + ni * 16 + n) * L_SEQ +
                              m0 + mi * 16 + 8 * half], acc[mi][ni]);
    }
}

// ---------------------------------------------------------------------------
// Kernel 4: flash-attention per (head, 64-query tile).
//   S = K^T Q / sqrt(E), causal, online softmax, B = V @ att.
//   block = 256 (8 waves), grid = (32, 8).
//   LDS S/P are query-major (128 keys contiguous per query).
// ---------------------------------------------------------------------------
__global__ void attention_kernel(const __bf16* __restrict__ Qt,   // L x 1024
                                 const __bf16* __restrict__ Kt,   // L x 1024
                                 const __bf16* __restrict__ Vr,   // 1024 x L
                                 __bf16* __restrict__ Bt) {       // L x 1024
    __shared__ float  S_ls[64 * 128];        // [query][key]
    __shared__ __bf16 P_ls[64 * 128];        // [query][key]
    __shared__ float  red[4 * 64];
    __shared__ float  m_ls[64], l_ls[64], alpha_ls[64];

    const int t = blockIdx.x;        // query tile (64 queries)
    const int h = blockIdx.y;        // head
    const int j0 = t * 64;
    const __bf16* Qh = Qt + h * E_DIM;                    // ld = 1024
    const __bf16* Kh = Kt + h * E_DIM;                    // ld = 1024
    const __bf16* Vh = Vr + (size_t)h * E_DIM * L_SEQ;    // ld = 2048

    const int lane = threadIdx.x & 31, w = threadIdx.x >> 5;
    const int half = lane >> 4, ln = lane & 15;
    const float rscale = 0.08838834764831845f;   // 1/sqrt(128)

    if (threadIdx.x < 64) {
        m_ls[threadIdx.x] = -__builtin_inff();
        l_ls[threadIdx.x] = 0.0f;
    }
    v8f acc[4] = {};
    __syncthreads();

    const int ntiles = t / 2 + 1;    // causal: 128-key tiles up to diagonal
    for (int kt = 0; kt < ntiles; ++kt) {
        const int i0 = kt * 128;

        // ---- phase 1: score tile S[64 q][128 key] -----------------------
        // wave w computes key rows [16w, 16w+16) x all 64 queries
#pragma unroll
        for (int nf = 0; nf < 4; ++nf) {
            v8f s = {};
#pragma unroll
            for (int e0 = 0; e0 < E_DIM; e0 += 32) {
                v16bf a = load_a_rm(Kh, EH, i0 + 16 * w, e0, lane);  // K^T
                v16bf b = load_b_t(Qh, EH, e0, j0 + nf * 16, lane);  // Q
                s = wmma_bf16(a, b, s);
            }
            float* sp = &S_ls[(nf * 16 + ln) * 128 + 16 * w + 8 * half];
#pragma unroll
            for (int r = 0; r < 8; ++r) sp[r] = s[r] * rscale;
        }
        __syncthreads();

        // ---- phase 2: online softmax, 4 partial-threads per query -------
        {
            const int q = threadIdx.x & 63;
            const int part = threadIdx.x >> 6;       // 0..3
            const int jg = j0 + q;
            int imax = jg - i0 + 1;                  // valid keys: i < imax
            if (imax > 128) imax = 128;
            const float* Sq = &S_ls[q * 128];
            float mx = -__builtin_inff();
            for (int i = part * 32; i < part * 32 + 32; ++i)
                if (i < imax) mx = fmaxf(mx, Sq[i]);
            red[part * 64 + q] = mx;
            __syncthreads();
            const float mold = m_ls[q];
            const float mnew = fmaxf(mold,
                fmaxf(fmaxf(red[q], red[64 + q]),
                      fmaxf(red[128 + q], red[192 + q])));
            float sum = 0.0f;
            __bf16* Pq = &P_ls[q * 128];
            for (int i = part * 32; i < part * 32 + 32; ++i) {
                float p = (i < imax) ? __expf(Sq[i] - mnew) : 0.0f;
                Pq[i] = (__bf16)p;
                sum += p;
            }
            __syncthreads();                 // red: write-after-read fence
            red[part * 64 + q] = sum;
            __syncthreads();
            if (part == 0) {
                float s4 = red[q] + red[64 + q] + red[128 + q] + red[192 + q];
                float alpha = __expf(mold - mnew);
                l_ls[q] = l_ls[q] * alpha + s4;
                m_ls[q] = mnew;
                alpha_ls[q] = alpha;
            }
        }
        __syncthreads();

        // ---- phase 3: rescale accumulators, B += V @ P ------------------
        // wave w owns embedding rows [16w, 16w+16)
#pragma unroll
        for (int nf = 0; nf < 4; ++nf) {
            const float al = alpha_ls[nf * 16 + ln];
#pragma unroll
            for (int r = 0; r < 8; ++r) acc[nf][r] *= al;
#pragma unroll
            for (int kk = 0; kk < 128; kk += 32) {
                v16bf a = load_a_rm(Vh, L_SEQ, 16 * w, i0 + kk, lane); // V
                v16bf b = load_b_t(P_ls, 128, kk, nf * 16, lane);      // P
                acc[nf] = wmma_bf16(a, b, acc[nf]);
            }
        }
        __syncthreads();
    }

    // ---- finalize: /= softmax sum, store B^T (L x 1024), e contiguous ---
#pragma unroll
    for (int nf = 0; nf < 4; ++nf) {
        const float inv = 1.0f / l_ls[nf * 16 + ln];
        v8f r = acc[nf];
#pragma unroll
        for (int i = 0; i < 8; ++i) r[i] *= inv;
        store_bf8(&Bt[(size_t)(j0 + nf * 16 + ln) * EH +
                      h * E_DIM + 16 * w + 8 * half], r);
    }
}

// ---------------------------------------------------------------------------
// Kernel 5: O = Head_agg(128x1024) @ B(1024x2048), B given transposed (Bt).
// Fused transpose+shift epilogue:
//   out1[(j+1)*128 + e] = O[e,j] for j<2047; out3[e] = O[e,2047].
// block = 256 (8 waves), grid = (2, 16)
// ---------------------------------------------------------------------------
__global__ void head_agg_kernel(const __bf16* __restrict__ Ha,   // 128 x 1024
                                const __bf16* __restrict__ Bt,   // L x 1024
                                float* __restrict__ out1,
                                float* __restrict__ out3) {
    const int lane = threadIdx.x & 31, w = threadIdx.x >> 5;
    const int m0 = blockIdx.x * 64 + (w >> 2) * 32;    // e row (0..127)
    const int n0 = blockIdx.y * 128 + (w & 3) * 32;    // seq (0..2047)

    v8f acc[2][2] = {};
    for (int k0 = 0; k0 < EH; k0 += 32) {
        if (k0 + 32 < EH) {
            __builtin_prefetch(&Ha[(size_t)m0 * EH + k0 + 32], 0, 1);
            __builtin_prefetch(&Bt[(size_t)n0 * EH + k0 + 32], 0, 1);
        }
        v16bf a0 = load_a_rm(Ha, EH, m0,      k0, lane);
        v16bf a1 = load_a_rm(Ha, EH, m0 + 16, k0, lane);
        v16bf b0 = load_b_t(Bt, EH, k0, n0,      lane);
        v16bf b1 = load_b_t(Bt, EH, k0, n0 + 16, lane);
        acc[0][0] = wmma_bf16(a0, b0, acc[0][0]);
        acc[0][1] = wmma_bf16(a0, b1, acc[0][1]);
        acc[1][0] = wmma_bf16(a1, b0, acc[1][0]);
        acc[1][1] = wmma_bf16(a1, b1, acc[1][1]);
    }
    const int half = lane >> 4, n = lane & 15;
#pragma unroll
    for (int mi = 0; mi < 2; ++mi)
#pragma unroll
        for (int ni = 0; ni < 2; ++ni) {
            const int j = n0 + ni * 16 + n;              // seq position
            const int mb = m0 + mi * 16 + 8 * half;      // first of 8 e-rows
            if (j < L_SEQ - 1) {
                float* dst = &out1[(size_t)(j + 1) * E_DIM + mb];
#pragma unroll
                for (int r = 0; r < 8; ++r) dst[r] = acc[mi][ni][r];
            } else {
#pragma unroll
                for (int r = 0; r < 8; ++r) out3[mb + r] = acc[mi][ni][r];
            }
        }
}

// ---------------------------------------------------------------------------
// Host-side launch
// ---------------------------------------------------------------------------
extern "C" void kernel_launch(void* const* d_in, const int* in_sizes, int n_in,
                              void* d_out, int out_size, void* d_ws, size_t ws_size,
                              hipStream_t stream) {
    (void)in_sizes; (void)n_in; (void)out_size; (void)ws_size;

    const int*   sq    = (const int*)d_in[0];
    const int*   cq    = (const int*)d_in[1];
    const float* item  = (const float*)d_in[2];
    const float* uinit = (const float*)d_in[3];
    const float* wrep  = (const float*)d_in[4];
    const float* Wq    = (const float*)d_in[5];
    const float* Wk    = (const float*)d_in[6];
    const float* Wv    = (const float*)d_in[7];
    const float* Hagg  = (const float*)d_in[8];

    float* out  = (float*)d_out;
    float* out1 = out;                        // u_state[:-1]  (2048 x 128)
    float* out2 = out + L_SEQ * E_DIM;        // I_q.T         (2048 x 128)
    float* out3 = out + 2 * L_SEQ * E_DIM;    // u_state[-1]   (128)

    // workspace carve-up (bf16 buffers; every block is 256B-aligned by size)
    char* ws = (char*)d_ws;
    __bf16* I_t   = (__bf16*)ws; ws += (size_t)L_SEQ * E_DIM * 2;   // L x 128
    __bf16* Wq_bf = (__bf16*)ws; ws += (size_t)EH * E_DIM * 2;      // 1024 x 128
    __bf16* Wk_bf = (__bf16*)ws; ws += (size_t)EH * E_DIM * 2;
    __bf16* Wv_bf = (__bf16*)ws; ws += (size_t)EH * E_DIM * 2;
    __bf16* Ha_bf = (__bf16*)ws; ws += (size_t)E_DIM * EH * 2;      // 128 x 1024
    __bf16* Qt    = (__bf16*)ws; ws += (size_t)L_SEQ * EH * 2;      // L x 1024
    __bf16* Kt    = (__bf16*)ws; ws += (size_t)L_SEQ * EH * 2;      // L x 1024
    __bf16* Vr    = (__bf16*)ws; ws += (size_t)EH * L_SEQ * 2;      // 1024 x L
    __bf16* Bt    = (__bf16*)ws; ws += (size_t)L_SEQ * EH * 2;      // L x 1024

    const int nW = EH * E_DIM;     // 131072 per weight matrix
    cvt_f32_bf16_kernel<<<(nW + 255) / 256, 256, 0, stream>>>(Wq, Wq_bf, nW);
    cvt_f32_bf16_kernel<<<(nW + 255) / 256, 256, 0, stream>>>(Wk, Wk_bf, nW);
    cvt_f32_bf16_kernel<<<(nW + 255) / 256, 256, 0, stream>>>(Wv, Wv_bf, nW);
    cvt_f32_bf16_kernel<<<(nW + 255) / 256, 256, 0, stream>>>(Hagg, Ha_bf, nW);

    gather_kernel<<<L_SEQ, E_DIM, 0, stream>>>(sq, cq, item, wrep, uinit,
                                               I_t, out1, out2);

    gemm_qkv_kernel<<<dim3(16, 16, 3), 256, 0, stream>>>(
        I_t, Wq_bf, Wk_bf, Wv_bf, Qt, Kt, Vr);

    attention_kernel<<<dim3(L_SEQ / 64, H_NUM), 256, 0, stream>>>(
        Qt, Kt, Vr, Bt);

    head_agg_kernel<<<dim3(E_DIM / 64, L_SEQ / 128), 256, 0, stream>>>(
        Ha_bf, Bt, out1, out3);
}
